// StructuralMambaMixer_41351945126097
// MI455X (gfx1250) — matrix operand order repaired and checked by reading, not compiled, not run on hardware
//
#include <hip/hip_runtime.h>

// ---------------- problem constants (from reference) ----------------
#define B_SZ    4
#define SEQ     1024
#define D_MODEL 1024
#define D_INNER 2048
#define D_STATE 16
#define D_CONV  4
#define DT_RANK 64
#define MROWS   (B_SZ * SEQ)          // 4096
#define NPROJ   (2 * D_INNER)         // 4096
#define NSSM    (DT_RANK + 2*D_STATE) // 96 (logical)
#define NSSMP   128                   // padded to block-tile multiple

typedef __attribute__((ext_vector_type(16))) __bf16 v16bf;
typedef __attribute__((ext_vector_type(8)))  float  v8f;

// ---------------- fragment helpers (CDNA5 WMMA VGPR layouts) ----------------
// A 16x32 bf16 (MxK): lanes 0-15 row m: elems[0..7]=K k0..+7, elems[8..15]=K k0+16..+23;
//                     lanes 16-31: K shifted +8.  => two 16B chunks at +0 / +16 elems.
// B 32x16 bf16 (KxN): lane n(0..15) holds column n, K order as A (k0..+7, k0+16..+23); lanes 16-31: K+8.
__device__ __forceinline__ v16bf load_frag_2x16(const __bf16* p0, const __bf16* p1) {
  union { uint4 q[2]; v16bf v; } u;
  u.q[0] = *reinterpret_cast<const uint4*>(p0);
  u.q[1] = *reinterpret_cast<const uint4*>(p1);
  return u.v;
}
__device__ __forceinline__ v8f zero_acc() {
  v8f a;
#pragma unroll
  for (int i = 0; i < 8; ++i) a[i] = 0.0f;
  return a;
}
// Flat representation of an LDS pointer: addr[31:0] == LDS offset (ISA 10.2 aperture map)
__device__ __forceinline__ unsigned lds_addr_of(const void* p) {
  return (unsigned)(unsigned long long)p;
}
// CDNA5 async copy global -> LDS (16B per lane), tracked by ASYNCcnt
__device__ __forceinline__ void stage_b128(unsigned ldsdst, const __bf16* g) {
  asm volatile("global_load_async_to_lds_b128 %0, %1, off" :: "v"(ldsdst), "v"(g) : "memory");
}
__device__ __forceinline__ void wait_asynccnt0() {
  asm volatile("s_wait_asynccnt 0x0" ::: "memory");
}

#define WMMA_BF16(a, b, c) \
  __builtin_amdgcn_wmma_f32_16x16x32_bf16(false, (a), false, (b), (short)0, (c), false, false)

// ---------------- C[M,N](f32) = A[M,K](bf16) * W[N,K](bf16)^T ----------------
// block: 256 thr = 8 waves as 4(M) x 2(N). block tile 128x128, wave tile 32x64, K-step 32,
// K loop unrolled by 2 => compile-time LDS double buffering. B tile (128x32) staged via
// async-to-LDS; A fragments software-pipelined one phase ahead.
// Requires: M % 128 == 0, N % 128 == 0 (pad), K % 64 == 0 or K == 64.
__launch_bounds__(256)
__global__ void gemm_xwt_bf16_wmma(const __bf16* __restrict__ A, const __bf16* __restrict__ W,
                                   float* __restrict__ C, int K, int lda, int ldb, int ldc) {
  __shared__ __attribute__((aligned(16))) __bf16 smemB[2][128][40]; // 80B rows: bank-spread

  const int lane    = threadIdx.x & 31;
  const int wave    = threadIdx.x >> 5;
  const int l15     = lane & 15;
  const int halfsel = lane >> 4;             // 0 or 1
  const int wm      = wave >> 1;             // 0..3
  const int wn      = wave & 1;              // 0..1
  const int m0      = blockIdx.y * 128 + wm * 32;
  const int gn0     = blockIdx.x * 128;

  // staging assignment: thread copies 16 contiguous bf16 of one B row per buffer
  const int srow = threadIdx.x >> 1;          // 0..127
  const int scol = (threadIdx.x & 1) * 16;    // 0 or 16
  const __bf16* gB = W + (size_t)(gn0 + srow) * ldb + scol;
  const unsigned ldsdst0 = lds_addr_of(&smemB[0][srow][scol]);
  const unsigned ldsdst1 = lds_addr_of(&smemB[1][srow][scol]);

  v8f accA0 = zero_acc(), accA1 = zero_acc(), accA2 = zero_acc(), accA3 = zero_acc();
  v8f accB0 = zero_acc(), accB1 = zero_acc(), accB2 = zero_acc(), accB3 = zero_acc();

  const __bf16* arow0 = A + (size_t)(m0 + l15)      * lda + halfsel * 8;
  const __bf16* arow1 = A + (size_t)(m0 + 16 + l15) * lda + halfsel * 8;

  // prologue: stage k=0 into buffer 0, load A fragments for k=0
  stage_b128(ldsdst0,      gB);
  stage_b128(ldsdst0 + 16, gB + 8);
  v16bf a0  = load_frag_2x16(arow0, arow0 + 16);
  v16bf a1  = load_frag_2x16(arow1, arow1 + 16);
  v16bf a0n = a0, a1n = a1;

#define LOAD_B_FRAGS(BUFIDX)                                                   \
  const __bf16* pbase = &smemB[(BUFIDX)][wn * 64 + l15][halfsel * 8];          \
  v16bf b0 = load_frag_2x16(pbase,           pbase + 16);                      \
  v16bf b1 = load_frag_2x16(pbase + 16 * 40, pbase + 16 * 40 + 16);            \
  v16bf b2 = load_frag_2x16(pbase + 32 * 40, pbase + 32 * 40 + 16);            \
  v16bf b3 = load_frag_2x16(pbase + 48 * 40, pbase + 48 * 40 + 16);

#define WMMA_ALL(AA0, AA1)                                                     \
  accA0 = WMMA_BF16(AA0, b0, accA0); accB0 = WMMA_BF16(AA1, b0, accB0);        \
  accA1 = WMMA_BF16(AA0, b1, accA1); accB1 = WMMA_BF16(AA1, b1, accB1);        \
  accA2 = WMMA_BF16(AA0, b2, accA2); accB2 = WMMA_BF16(AA1, b2, accB2);        \
  accA3 = WMMA_BF16(AA0, b3, accA3); accB3 = WMMA_BF16(AA1, b3, accB3);

  for (int k0 = 0; k0 < K; k0 += 64) {
    // ---- phase 0: consume buffer 0 (k = k0); stream k0+32 into buffer 1 ----
    wait_asynccnt0();           // this thread's stage into buf0 landed
    __syncthreads();            // all stages into buf0 visible
    {
      const __bf16* g = gB + (k0 + 32);          // k0+32 < K always (K % 64 == 0)
      stage_b128(ldsdst1,      g);
      stage_b128(ldsdst1 + 16, g + 8);
      const __bf16* pn0 = arow0 + k0 + 32;
      const __bf16* pn1 = arow1 + k0 + 32;
      a0n = load_frag_2x16(pn0, pn0 + 16);
      a1n = load_frag_2x16(pn1, pn1 + 16);
      __builtin_prefetch(pn0 + 32, 0, 3);        // global_prefetch_b8 next A tile
    }
    {
      LOAD_B_FRAGS(0)
      WMMA_ALL(a0, a1)
    }
    __syncthreads();            // reads of buf0 done before it is restaged

    // ---- phase 1: consume buffer 1 (k = k0+32); stream k0+64 into buffer 0 ----
    wait_asynccnt0();
    __syncthreads();
    if (k0 + 64 < K) {
      const __bf16* g = gB + (k0 + 64);
      stage_b128(ldsdst0,      g);
      stage_b128(ldsdst0 + 16, g + 8);
      const __bf16* pn0 = arow0 + k0 + 64;
      const __bf16* pn1 = arow1 + k0 + 64;
      a0 = load_frag_2x16(pn0, pn0 + 16);
      a1 = load_frag_2x16(pn1, pn1 + 16);
    }
    {
      LOAD_B_FRAGS(1)
      WMMA_ALL(a0n, a1n)
    }
    __syncthreads();
  }
#undef LOAD_B_FRAGS
#undef WMMA_ALL

  // C/D layout: VGPR v: lanes0-15 -> M=base+v, lanes16-31 -> M=base+v+8; N = col
#define STORE_TILE(acc, mbase, j)                                          \
  {                                                                        \
    int col = gn0 + wn * 64 + (j) * 16 + l15;                              \
    _Pragma("unroll")                                                      \
    for (int v = 0; v < 8; ++v)                                            \
      C[(size_t)((mbase) + v + halfsel * 8) * ldc + col] = (acc)[v];       \
  }
  STORE_TILE(accA0, m0, 0)      STORE_TILE(accA1, m0, 1)
  STORE_TILE(accA2, m0, 2)      STORE_TILE(accA3, m0, 3)
  STORE_TILE(accB0, m0 + 16, 0) STORE_TILE(accB1, m0 + 16, 1)
  STORE_TILE(accB2, m0 + 16, 2) STORE_TILE(accB3, m0 + 16, 3)
#undef STORE_TILE
}

// ---------------- elementwise / fused kernels ----------------
__global__ void cvt_f32_bf16(const float* __restrict__ in, __bf16* __restrict__ out, int n) {
  int i = blockIdx.x * blockDim.x + threadIdx.x;
  if (i < n) out[i] = (__bf16)in[i];
}

// depthwise causal conv1d (d_conv=4) + bias + silu, reading x = proj[:, :D_INNER]
__global__ void conv_silu(const float* __restrict__ proj, const float* __restrict__ cw,
                          const float* __restrict__ cb, __bf16* __restrict__ u, int total) {
  int i = blockIdx.x * blockDim.x + threadIdx.x;
  if (i >= total) return;
  int c = i & (D_INNER - 1);
  int t = (i >> 11) & (SEQ - 1);
  int b = i >> 21;
  float acc = cb[c];
#pragma unroll
  for (int k = 0; k < D_CONV; ++k) {
    int tt = t - (D_CONV - 1) + k;
    if (tt >= 0) acc += proj[(size_t)(b * SEQ + tt) * NPROJ + c] * cw[c * D_CONV + k];
  }
  float s = acc / (1.0f + __expf(-acc));
  u[i] = (__bf16)s;
}

// dt_low = ssm[:, :DT_RANK] as bf16 GEMM operand (ssm row stride = NSSMP)
__global__ void extract_dtlow(const float* __restrict__ ssm, __bf16* __restrict__ dtlow, int total) {
  int i = blockIdx.x * blockDim.x + threadIdx.x;
  if (i >= total) return;
  int r = i >> 6, j = i & (DT_RANK - 1);
  dtlow[i] = (__bf16)ssm[(size_t)r * NSSMP + j];
}

// dt = softplus(dt_raw + b_dt + time_scale * clip(td, 0, 100))   (in place)
__global__ void dt_softplus(float* __restrict__ dt, const float* __restrict__ b_dt,
                            const float* __restrict__ ts, const float* __restrict__ td, int total) {
  int i = blockIdx.x * blockDim.x + threadIdx.x;
  if (i >= total) return;
  int c  = i & (D_INNER - 1);
  int rt = i >> 11;                 // b*SEQ + t
  float tdv = td[rt];
  tdv = fminf(fmaxf(tdv, 0.0f), 100.0f);
  float v = dt[i] + b_dt[c] + ts[c] * tdv;
  dt[i] = (v > 20.0f) ? v : log1pf(__expf(v));
}

// selective scan: one lane per (batch, channel); h[16] in registers;
// fuses +u*D and *silu(gate); emits bf16 y for the out-proj WMMA GEMM.
__launch_bounds__(256)
__global__ void ssm_scan(const float* __restrict__ dt, const __bf16* __restrict__ u,
                         const float* __restrict__ ssm, const float* __restrict__ proj,
                         const float* __restrict__ A_log, const float* __restrict__ Dp,
                         __bf16* __restrict__ y) {
  int tid = blockIdx.x * blockDim.x + threadIdx.x;   // 0 .. B_SZ*D_INNER-1
  int b = tid >> 11;
  int c = tid & (D_INNER - 1);
  float As[D_STATE];
#pragma unroll
  for (int s = 0; s < D_STATE; ++s) As[s] = -__expf(A_log[c * D_STATE + s]);
  const float Dv = Dp[c];
  float h[D_STATE] = {};
  for (int t = 0; t < SEQ; ++t) {
    size_t base = (size_t)(b * SEQ + t);
    if (t + 8 < SEQ) {                 // prefetch ahead: scan runs at ~1 wave/SIMD
      size_t pf = base + 8;
      __builtin_prefetch(&dt[pf * D_INNER + c], 0, 3);
      __builtin_prefetch(&u[pf * D_INNER + c], 0, 3);
      __builtin_prefetch(&proj[pf * NPROJ + D_INNER + c], 0, 3);
    }
    float dtv = dt[base * D_INNER + c];
    float uv  = (float)u[base * D_INNER + c];
    const float* bc = ssm + base * NSSMP + DT_RANK;  // Bp[0..15], Cp[16..31]
    float yv = 0.0f;
#pragma unroll
    for (int s = 0; s < D_STATE; ++s) {
      float dA = __expf(dtv * As[s]);
      h[s] = dA * h[s] + dtv * bc[s] * uv;
      yv += h[s] * bc[D_STATE + s];
    }
    float g  = proj[base * NPROJ + D_INNER + c];
    float sg = g / (1.0f + __expf(-g));
    y[base * D_INNER + c] = (__bf16)((yv + uv * Dv) * sg);
  }
}

// ---------------- host launcher ----------------
extern "C" void kernel_launch(void* const* d_in, const int* in_sizes, int n_in,
                              void* d_out, int out_size, void* d_ws, size_t ws_size,
                              hipStream_t stream) {
  (void)in_sizes; (void)n_in; (void)out_size; (void)ws_size;
  const float* hs   = (const float*)d_in[0];
  const float* td   = (const float*)d_in[1];
  const float* Win  = (const float*)d_in[2];
  const float* cw   = (const float*)d_in[3];
  const float* cb   = (const float*)d_in[4];
  const float* Wx   = (const float*)d_in[5];
  const float* Wdt  = (const float*)d_in[6];
  const float* bdt  = (const float*)d_in[7];
  const float* ts   = (const float*)d_in[8];
  const float* Alog = (const float*)d_in[9];
  const float* Dp   = (const float*)d_in[10];
  const float* Wout = (const float*)d_in[11];
  float* out = (float*)d_out;

  char* ws = (char*)d_ws;
  size_t off = 0;
  auto salloc = [&](size_t bytes) {
    char* p = ws + off;
    off = (off + bytes + 255) & ~(size_t)255;
    return p;
  };

  __bf16* hs_bf   = (__bf16*)salloc((size_t)MROWS * D_MODEL * 2);
  __bf16* Win_bf  = (__bf16*)salloc((size_t)NPROJ * D_MODEL * 2);
  __bf16* Wx_bf   = (__bf16*)salloc((size_t)NSSMP * D_INNER * 2);   // rows 96..127 pad (never read downstream)
  __bf16* Wdt_bf  = (__bf16*)salloc((size_t)D_INNER * DT_RANK * 2);
  __bf16* Wout_bf = (__bf16*)salloc((size_t)D_MODEL * D_INNER * 2);
  float*  proj    = (float*)salloc((size_t)MROWS * NPROJ * 4);      // [x | gate]
  __bf16* u_bf    = (__bf16*)salloc((size_t)MROWS * D_INNER * 2);
  float*  ssm     = (float*)salloc((size_t)MROWS * NSSMP * 4);      // [dt_low | B | C | pad]
  __bf16* dtlo_bf = (__bf16*)salloc((size_t)MROWS * DT_RANK * 2);
  float*  dtbuf   = (float*)salloc((size_t)MROWS * D_INNER * 4);
  __bf16* y_bf    = (__bf16*)salloc((size_t)MROWS * D_INNER * 2);

  const int TPB = 256;
  auto cdiv = [](int a, int b) { return (a + b - 1) / b; };

  // 1) weight / activation conversions to bf16
  { int n = MROWS * D_MODEL;   cvt_f32_bf16<<<cdiv(n,TPB),TPB,0,stream>>>(hs,   hs_bf,   n); }
  { int n = NPROJ * D_MODEL;   cvt_f32_bf16<<<cdiv(n,TPB),TPB,0,stream>>>(Win,  Win_bf,  n); }
  { int n = NSSM * D_INNER;    cvt_f32_bf16<<<cdiv(n,TPB),TPB,0,stream>>>(Wx,   Wx_bf,   n); }
  { int n = D_INNER * DT_RANK; cvt_f32_bf16<<<cdiv(n,TPB),TPB,0,stream>>>(Wdt,  Wdt_bf,  n); }
  { int n = D_MODEL * D_INNER; cvt_f32_bf16<<<cdiv(n,TPB),TPB,0,stream>>>(Wout, Wout_bf, n); }

  // 2) in_proj: proj = hs @ W_in^T   (4096 x 4096, K=1024)
  gemm_xwt_bf16_wmma<<<dim3(NPROJ/128, MROWS/128), TPB, 0, stream>>>(
      hs_bf, Win_bf, proj, D_MODEL, D_MODEL, D_MODEL, NPROJ);

  // 3) depthwise causal conv + silu -> u (bf16)
  { int n = MROWS * D_INNER; conv_silu<<<cdiv(n,TPB),TPB,0,stream>>>(proj, cw, cb, u_bf, n); }

  // 4) x_proj: ssm = u @ W_x^T   (4096 x 128pad, K=2048)
  gemm_xwt_bf16_wmma<<<dim3(NSSMP/128, MROWS/128), TPB, 0, stream>>>(
      u_bf, Wx_bf, ssm, D_INNER, D_INNER, D_INNER, NSSMP);

  // 5) dt_proj: dt_raw = dt_low @ W_dt^T   (4096 x 2048, K=64)
  { int n = MROWS * DT_RANK; extract_dtlow<<<cdiv(n,TPB),TPB,0,stream>>>(ssm, dtlo_bf, n); }
  gemm_xwt_bf16_wmma<<<dim3(D_INNER/128, MROWS/128), TPB, 0, stream>>>(
      dtlo_bf, Wdt_bf, dtbuf, DT_RANK, DT_RANK, DT_RANK, D_INNER);

  // 6) dt = softplus(dt_raw + b_dt + time_scale * clip(td))
  { int n = MROWS * D_INNER; dt_softplus<<<cdiv(n,TPB),TPB,0,stream>>>(dtbuf, bdt, ts, td, n); }

  // 7) selective scan (+ skip D, * silu(gate)) -> y (bf16)
  ssm_scan<<<(B_SZ * D_INNER) / TPB, TPB, 0, stream>>>(dtbuf, u_bf, ssm, proj, Alog, Dp, y_bf);

  // 8) out_proj: out = y @ W_out^T   (4096 x 1024, K=2048), fp32 to d_out
  gemm_xwt_bf16_wmma<<<dim3(D_MODEL/128, MROWS/128), TPB, 0, stream>>>(
      y_bf, Wout_bf, out, D_INNER, D_INNER, D_INNER, D_MODEL);
}
